// LayoutPlanner_76828374991758
// MI455X (gfx1250) — matrix operand order repaired and checked by reading, not compiled, not run on hardware
//
#include <hip/hip_runtime.h>

typedef __attribute__((ext_vector_type(16))) _Float16 v16h;
typedef __attribute__((ext_vector_type(8)))  _Float16 v8h;
typedef __attribute__((ext_vector_type(4)))  _Float16 v4h;
typedef __attribute__((ext_vector_type(8)))  float    v8f;
typedef __attribute__((ext_vector_type(4)))  float    f4;

#define BB 8
#define SS 4096
#define DD 768
#define NE 64
#define K2 1536           // 2*D
#define ROWPAD 1544       // 1536 + 8 halves (bank-conflict pad, keeps 16B alignment)

// ---------------------------------------------------------------------------
// Kernel 0: build combined transposed weight panel Wcat^T (32 x 1536, f16)
//   rows  0..7  : panel_w  (768x8), K>=768 zero
//   rows  8..15 : dialog_w (1536x8)
//   rows 16..19 : char_w   (1536x4)
//   rows 20..31 : zero
// plus fused bias vector (32 floats).
// ---------------------------------------------------------------------------
__global__ void build_wt_kernel(const float* __restrict__ panel_w,
                                const float* __restrict__ dialog_w,
                                const float* __restrict__ char_w,
                                const float* __restrict__ panel_b,
                                const float* __restrict__ dialog_b,
                                const float* __restrict__ char_b,
                                _Float16* __restrict__ wt,
                                float* __restrict__ biasCat) {
    int idx = blockIdx.x * 256 + threadIdx.x;      // 32*1536 = 49152 elems
    if (idx < 32 * K2) {
        int n = idx / K2, k = idx % K2;
        float v = 0.f;
        if (n < 8)       { if (k < DD) v = panel_w[k * 8 + n]; }
        else if (n < 16) { v = dialog_w[k * 8 + (n - 8)]; }
        else if (n < 20) { v = char_w[k * 4 + (n - 16)]; }
        wt[n * K2 + k] = (_Float16)v;
    }
    if (blockIdx.x == 0 && threadIdx.x < 32) {
        int n = threadIdx.x; float bv = 0.f;
        if (n < 8) bv = panel_b[n];
        else if (n < 16) bv = dialog_b[n - 8];
        else if (n < 20) bv = char_b[n - 16];
        biasCat[n] = bv;
    }
}

// ---------------------------------------------------------------------------
// Kernel 1: per-batch remap: elem_index -> sequence position of LAST panel
// token carrying that index (== reference scatter-max on monotonic 'kept',
// composed with pos_table).  -1 if no such panel.
// ---------------------------------------------------------------------------
__global__ void build_remap_kernel(const int* __restrict__ types,
                                   const int* __restrict__ eidx,
                                   int* __restrict__ remap) {
    __shared__ int r[NE];
    int b = blockIdx.x;
    if (threadIdx.x < NE) r[threadIdx.x] = -1;
    __syncthreads();
    for (int s = threadIdx.x; s < SS; s += blockDim.x) {
        if (types[b * SS + s] == 1)                     // TYPE_PANEL
            atomicMax(&r[eidx[b * SS + s]], s);         // ds_max_i32; max s == last
    }
    __syncthreads();
    if (threadIdx.x < NE) remap[b * NE + threadIdx.x] = r[threadIdx.x];
}

// ---------------------------------------------------------------------------
// Main kernel: one wave32 per 16-token tile.
//   lanes 0..15  : LayerNorm of own token   -> cat[tok][0..767]
//   lanes 16..31 : LayerNorm of parent token (recomputed from parent indices,
//                  avoiding a 100MB feats buffer) -> cat[tok][768..1535]
//   then 48 K-steps x 2 N-tiles of v_wmma_f32_16x16x32_f16 for all 3 heads.
// ---------------------------------------------------------------------------
__launch_bounds__(32)
__global__ void layout_main_kernel(
    const int*   __restrict__ types, const int* __restrict__ eidx,
    const int*   __restrict__ pidx,  const float* __restrict__ style,
    const float* __restrict__ type_emb,  const float* __restrict__ index_emb,
    const float* __restrict__ parent_emb, const float* __restrict__ style_w,
    const float* __restrict__ style_b,   const float* __restrict__ ln_g,
    const float* __restrict__ ln_b,      const _Float16* __restrict__ wt,
    const float* __restrict__ biasCat,   const int* __restrict__ remap,
    float* __restrict__ out) {

    __shared__ __align__(16) _Float16 cat[16][ROWPAD];  // 16 x 1536 f16 (+pad)
    __shared__ __align__(16) float sp[DD];              // style projection
    __shared__ int typeS[16], validS[16], pseqS[16];

    const int lane = threadIdx.x;
    const int bb   = blockIdx.y;
    const int s0   = blockIdx.x * 16;

    // style projection (per batch), once per block
    {
        float a0 = style[bb*4+0], a1 = style[bb*4+1];
        float a2 = style[bb*4+2], a3 = style[bb*4+3];
        for (int d = lane; d < DD; d += 32)
            sp[d] = a0*style_w[d] + a1*style_w[DD+d] + a2*style_w[2*DD+d]
                  + a3*style_w[3*DD+d] + style_b[d];
    }

    // token metadata + parent resolution
    if (lane < 16) {
        int s = s0 + lane;
        int t = types[bb*SS + s];
        typeS[lane] = t;
        int v = 0, ps = 0;
        if (t == 2 || t == 3) {                         // DIALOG or CHAR
            int pr = remap[bb*NE + pidx[bb*SS + s]];
            if (pr >= 0) { v = 1; ps = pr; }
        }
        validS[lane] = v; pseqS[lane] = ps;
    }
    __syncthreads();

    // one LN row per lane (recompute-x two-pass; embedding tables are L2-hot)
    {
        int tok = lane & 15;
        bool isParent = lane >= 16;
        bool active = true;
        int srow;
        if (isParent) { active = validS[tok] != 0; srow = pseqS[tok]; }
        else          { srow = s0 + tok; }
        int colBase = isParent ? DD : 0;

        if (active) {
            int t  = types[bb*SS + srow];
            int ei = eidx [bb*SS + srow];
            int pp = pidx [bb*SS + srow];
            const f4* te = (const f4*)(type_emb   + t  * DD);
            const f4* ie = (const f4*)(index_emb  + ei * DD);
            const f4* pe = (const f4*)(parent_emb + pp * DD);
            const f4* sv = (const f4*)sp;
            float sum = 0.f, sq = 0.f;
            for (int d4 = 0; d4 < DD/4; ++d4) {
                f4 x = te[d4] + ie[d4] + pe[d4] + sv[d4];
                sum += x.x + x.y + x.z + x.w;
                sq  += x.x*x.x + x.y*x.y + x.z*x.z + x.w*x.w;
            }
            float mu = sum * (1.f/DD);
            float var = sq * (1.f/DD) - mu*mu;
            float rs = rsqrtf(var + 1e-5f);
            const f4* g4 = (const f4*)ln_g;
            const f4* b4 = (const f4*)ln_b;
            for (int d4 = 0; d4 < DD/4; ++d4) {
                f4 x = te[d4] + ie[d4] + pe[d4] + sv[d4];
                f4 g = g4[d4], bo = b4[d4];
                v4h y;
                y[0] = (_Float16)((x.x - mu) * rs * g.x + bo.x);
                y[1] = (_Float16)((x.y - mu) * rs * g.y + bo.y);
                y[2] = (_Float16)((x.z - mu) * rs * g.z + bo.z);
                y[3] = (_Float16)((x.w - mu) * rs * g.w + bo.w);
                *(v4h*)&cat[tok][colBase + d4*4] = y;
            }
        } else {                                        // invalid parent -> zeros
            v4h z = {};
            for (int d4 = 0; d4 < DD/4; ++d4)
                *(v4h*)&cat[tok][colBase + d4*4] = z;
        }
    }
    __syncthreads();

    // ---- WMMA: cat(16x1536) @ WcatT rows -> two 16x16 f32 tiles -------------
    const int n = lane & 15;        // output column (and A row m — same lane id)
    const int g = lane >> 4;        // lane half-group
    v8f acc0 = {}, acc1 = {};
    #pragma unroll 4
    for (int kb = 0; kb < 48; ++kb) {
        int k0 = kb * 32;
        // A fragment: row m=n of cat; lane holds K in [g*16, g*16+16)
        // (same order as B)  -> two aligned 16B LDS reads
        v8h alo = *(const v8h*)&cat[n][k0 + g*16];
        v8h ahi = *(const v8h*)&cat[n][k0 + g*16 + 8];
        v16h a = __builtin_shufflevector(alo, ahi,
                 0,1,2,3,4,5,6,7,8,9,10,11,12,13,14,15);
        // B fragment: column n of Wcat == row n of WcatT; lane holds K in
        // [g*16, g*16+16) -> one contiguous 32B global (L2) read
        v16h b0 = *(const v16h*)(wt + n        * K2 + k0 + g*16);
        v16h b1 = *(const v16h*)(wt + (16 + n) * K2 + k0 + g*16);
        acc0 = __builtin_amdgcn_wmma_f32_16x16x32_f16(false, a, false, b0,
                                                      (short)0, acc0, false, false);
        acc1 = __builtin_amdgcn_wmma_f32_16x16x32_f16(false, a, false, b1,
                                                      (short)0, acc1, false, false);
    }

    // ---- bias + per-head masks + scatter to the three outputs ---------------
    float* panelO  = out;
    float* dialogO = out + (long)BB*SS*8;
    float* charO   = out + 2L*BB*SS*8;
    float bz0 = biasCat[n];
    float bz1 = biasCat[16 + n];
    #pragma unroll
    for (int r = 0; r < 8; ++r) {
        int m = g*8 + r;                       // token within tile (C/D layout)
        int t = typeS[m], vld = validS[m];
        long base = (long)(bb*SS + s0 + m);
        if (n < 8)  panelO [base*8 + n]     = (t == 1)        ? acc0[r] + bz0 : 0.f;
        else        dialogO[base*8 + (n-8)] = (t == 2 && vld) ? acc0[r] + bz0 : 0.f;
        if (n < 4)  charO  [base*4 + n]     = (t == 3 && vld) ? acc1[r] + bz1 : 0.f;
    }
}

// ---------------------------------------------------------------------------
extern "C" void kernel_launch(void* const* d_in, const int* in_sizes, int n_in,
                              void* d_out, int out_size, void* d_ws, size_t ws_size,
                              hipStream_t stream) {
    const int*   types    = (const int*)  d_in[0];
    const int*   eidx     = (const int*)  d_in[1];
    const int*   pidx     = (const int*)  d_in[2];
    const float* style    = (const float*)d_in[3];
    const float* type_emb = (const float*)d_in[4];
    const float* index_emb= (const float*)d_in[5];
    const float* parent_emb=(const float*)d_in[6];
    const float* style_w  = (const float*)d_in[7];
    const float* style_b  = (const float*)d_in[8];
    const float* ln_g     = (const float*)d_in[9];
    const float* ln_b     = (const float*)d_in[10];
    const float* panel_w  = (const float*)d_in[11];
    const float* panel_b  = (const float*)d_in[12];
    const float* dialog_w = (const float*)d_in[13];
    const float* dialog_b = (const float*)d_in[14];
    const float* char_w   = (const float*)d_in[15];
    const float* char_b   = (const float*)d_in[16];

    // workspace: WcatT f16 (98304 B) | biasCat f32 (128 B) | remap int (2048 B)
    _Float16* wt      = (_Float16*)d_ws;
    float*    biasCat = (float*)((char*)d_ws + 32 * K2 * 2);
    int*      remap   = (int*)  ((char*)d_ws + 32 * K2 * 2 + 128);

    build_wt_kernel<<<(32 * K2 + 255) / 256, 256, 0, stream>>>(
        panel_w, dialog_w, char_w, panel_b, dialog_b, char_b, wt, biasCat);

    build_remap_kernel<<<BB, 256, 0, stream>>>(types, eidx, remap);

    layout_main_kernel<<<dim3(SS / 16, BB), 32, 0, stream>>>(
        types, eidx, pidx, style, type_emb, index_emb, parent_emb,
        style_w, style_b, ln_g, ln_b, wt, biasCat, remap, (float*)d_out);
}
